// GNNModel_58841051955383
// MI455X (gfx1250) — matrix operand order repaired and checked by reading, not compiled, not run on hardware
//
#include <hip/hip_runtime.h>

#define B_    4
#define N_    1024
#define T_    16
#define FIN_  32
#define H_    128
#define LH_   64
#define BTN_  (B_ * T_ * N_)     /* 65536 rows */
#define NEGV  (-9e15f)
#define ALPHA_ 0.2f

typedef __attribute__((ext_vector_type(16))) __bf16 v16bf;
typedef __attribute__((ext_vector_type(8)))  float  v8f;

union Frag   { v16bf v; uint4 q[2]; };          // 32 bytes
union BfPair { unsigned int u; __bf16 h[2]; };

// ---------------- WMMA fragment index helpers (ISA 7.12.2, wave32) -----------
// 16-bit A 16x32: lane L: row=L%16; halfs at Kbase+{0..7},{16..23}, Kbase=(L>=16?8:0)
// 16-bit B 32x16: lane L: col=L%16; halfs at Kbase+{0..7},{16..23}, Kbase=(L>=16?8:0)
// f32 C/D 16x16: element i of lane L -> row = i + (L>=16?8:0), col = L%16
__device__ __forceinline__ int kb_idx(int e, int lane) {
  return ((lane >= 16) ? 8 : 0) + ((e < 8) ? e : e + 8);
}

__device__ __forceinline__ float sigmf(float x) { return 1.f / (1.f + __expf(-x)); }
__device__ __forceinline__ float eluf(float x)  { return (x > 0.f) ? x : (__expf(x) - 1.f); }
__device__ __forceinline__ float leakyf(float x){ return (x > 0.f) ? x : (ALPHA_ * x); }

// ---------------- pack a weight matrix into B-fragment order -----------------
__global__ void k_pack_b(const float* __restrict__ in, __bf16* __restrict__ out,
                         int Kb, int Nb, int ldin, int trans) {
  int o = blockIdx.x * 256 + threadIdx.x;
  if (o >= Kb * Nb) return;
  int e    = o & 15;
  int lane = (o >> 4) & 31;
  int tile = o >> 9;
  int tiles_n = Nb >> 4;
  int ks = tile / tiles_n, tn = tile % tiles_n;
  int k = ks * 32 + kb_idx(e, lane);
  int c = tn * 16 + (lane & 15);
  float v = trans ? in[(long)c * ldin + k] : in[(long)k * ldin + c];
  out[o] = (__bf16)v;
}

// ---------------- GEMM:  WhT = (A * B)^T,  B pre-packed ----------------------
// MODE 0: A = x (f32, layout (B,N,T,FIN));  MODE 1: A = bf16 contiguous rows.
// Each wave computes TWO adjacent 16x16 tiles (shared A fragment).
// Output transposed per (b,t) slice: CoutT[(bt*H + c)*N + i],  r = bt*N + i
template <int MODE, int K>
__global__ void __launch_bounds__(256)
k_gemm(const void* __restrict__ Aptr, const __bf16* __restrict__ Bp,
       __bf16* __restrict__ CoutT) {
  int lane = threadIdx.x & 31;
  int w    = blockIdx.x * 8 + (threadIdx.x >> 5);
  int tm = w >> 2, tp = w & 3;
  int tn0 = tp * 2, tn1 = tn0 + 1;               // tiles_n = H/16 = 8
  int row = tm * 16 + (lane & 15);

  long arow;
  if (MODE == 0) {
    int b = row >> 14, rem = row & 16383;
    int t = rem >> 10, n = rem & 1023;
    arow = (((long)(b * N_ + n)) * T_ + t) * FIN_;
  } else {
    arow = (long)row * K;
  }
  int klo = (lane >= 16) ? 8 : 0;

  v8f acc0 = {}, acc1 = {};
#pragma unroll
  for (int k0 = 0; k0 < K; k0 += 32) {
    Frag fa, f0, f1;
    if (MODE == 0) {
      const float4* f4 = (const float4*)((const float*)Aptr + arow + k0 + klo);
      float4 u0 = f4[0], u1 = f4[1], u2 = f4[4], u3 = f4[5];
      float t16[16] = {u0.x,u0.y,u0.z,u0.w, u1.x,u1.y,u1.z,u1.w,
                       u2.x,u2.y,u2.z,u2.w, u3.x,u3.y,u3.z,u3.w};
#pragma unroll
      for (int e = 0; e < 16; ++e) fa.v[e] = (__bf16)t16[e];
    } else {
      const uint4* a4 = (const uint4*)((const __bf16*)Aptr + arow + k0 + klo);
      fa.q[0] = a4[0];
      fa.q[1] = a4[2];                            // +16 halfs
    }
    const uint4* b0 = (const uint4*)(Bp + (((long)(k0 >> 5) * 8 + tn0) << 9) + (lane << 4));
    f0.q[0] = b0[0]; f0.q[1] = b0[1];
    f1.q[0] = b0[32]; f1.q[1] = b0[33];           // +256 halfs (32 x uint4)
    acc0 = __builtin_amdgcn_wmma_f32_16x16x32_bf16(false, fa.v, false, f0.v, (short)0, acc0, false, false);
    acc1 = __builtin_amdgcn_wmma_f32_16x16x32_bf16(false, fa.v, false, f1.v, (short)0, acc1, false, false);
  }

  int bt = (tm * 16) >> 10;                       // tile never crosses a (b,t) slice
  int c0 = tn0 * 16 + (lane & 15);
#pragma unroll
  for (int i = 0; i < 8; ++i) {
    int ii = (tm * 16 + i + ((lane >= 16) ? 8 : 0)) & 1023;
    CoutT[((long)bt * H_ + c0) * N_ + ii]      = (__bf16)acc0[i];
    CoutT[((long)bt * H_ + c0 + 16) * N_ + ii] = (__bf16)acc1[i];
  }
}

// ---------------- attention rank-1 scores (WhT layout) -----------------------
__global__ void k_scores(const __bf16* __restrict__ WhT, const float* __restrict__ a,
                         float* __restrict__ ssrc, float* __restrict__ sdst, int rows) {
  int r = blockIdx.x * 256 + threadIdx.x;
  if (r >= rows) return;
  const __bf16* base = WhT + (long)(r >> 10) * H_ * N_ + (r & 1023);
  float s0 = 0.f, s1 = 0.f;
#pragma unroll 4
  for (int k = 0; k < H_; ++k) {
    float v = (float)base[(long)k * N_];
    s0 += v * a[k];
    s1 += v * a[H_ + k];
  }
  ssrc[r] = s0;
  sdst[r] = s1;
}

// ---------------- softmax row stats, one wave per row, branchless ------------
__global__ void k_stats(const int* __restrict__ adj, const float* __restrict__ ssrc,
                        const float* __restrict__ sdst,
                        float* __restrict__ rmax, float* __restrict__ rsum) {
  int lane = threadIdx.x & 31;
  int wrow = blockIdx.x * 8 + (threadIdx.x >> 5);   // ((b*T)+t)*N + i
  int b = wrow >> 14;
  int i = wrow & 1023;
  const int4*   a4 = (const int4*)(adj + ((long)b * N_ + i) * N_);
  const float4* s4 = (const float4*)(sdst + (long)(wrow >> 10) * N_);
  float si = ssrc[wrow];

  float ev[32];
  float m = -3.4e38f;
#pragma unroll
  for (int c = 0; c < 8; ++c) {
    int q = c * 32 + lane;
    int4   av = a4[q];
    float4 sv = s4[q];
    float e0 = (av.x > 0) ? leakyf(si + sv.x) : NEGV;
    float e1 = (av.y > 0) ? leakyf(si + sv.y) : NEGV;
    float e2 = (av.z > 0) ? leakyf(si + sv.z) : NEGV;
    float e3 = (av.w > 0) ? leakyf(si + sv.w) : NEGV;
    ev[c * 4 + 0] = e0; ev[c * 4 + 1] = e1; ev[c * 4 + 2] = e2; ev[c * 4 + 3] = e3;
    m = fmaxf(m, fmaxf(fmaxf(e0, e1), fmaxf(e2, e3)));
  }
#pragma unroll
  for (int off = 16; off; off >>= 1) m = fmaxf(m, __shfl_xor(m, off, 32));

  float s = 0.f;
#pragma unroll
  for (int c = 0; c < 32; ++c) s += __expf(ev[c] - m);
#pragma unroll
  for (int off = 16; off; off >>= 1) s += __shfl_xor(s, off, 32);

  if (lane == 0) { rmax[wrow] = m; rsum[wrow] = s; }
}

// ---------------- attention GEMM: out = elu(elu(softmax(e) @ Wh)) ------------
// WhT: per (b,t) slice H x N (transposed).  Output row-major (B,T,N,H) bf16.
// 8 waves; wave wv owns output cols [16*wv, 16*wv+16).  Double-buffered A tile.
__global__ void __launch_bounds__(256)
k_attn(const __bf16* __restrict__ WhT, const int* __restrict__ adj,
       const float* __restrict__ ssrc, const float* __restrict__ sdst,
       const float* __restrict__ rmax, const float* __restrict__ rsum,
       __bf16* __restrict__ Hout) {
  __shared__ __align__(16) unsigned int pFragU[2][16 * 16];  // 2 x 512 bf16
  __shared__ float sRow[16], mRow[16], rInv[16];

  int bt = blockIdx.x >> 6;
  int i0 = (blockIdx.x & 63) * 16;
  int b  = bt >> 4;
  int tid = threadIdx.x, lane = tid & 31, wv = tid >> 5;
  long rowbase = (long)bt * N_;

  if (tid < 16) {
    sRow[tid] = ssrc[rowbase + i0 + tid];
    mRow[tid] = rmax[rowbase + i0 + tid];
    rInv[tid] = 1.f / rsum[rowbase + i0 + tid];
  }
  __syncthreads();

  const float*  sd    = sdst + rowbase;
  const __bf16* whSlc = WhT + (long)bt * H_ * N_;
  int rr = tid >> 4;                 // builder row
  int jj = (tid << 1) & 31;          // builder K pair (even)
  int laneT = rr + ((jj & 8) ? 16 : 0);
  int vT    = ((jj & 16) ? 4 : 0) + ((jj >> 1) & 3);
  const int* adjRow = adj + ((long)b * N_ + i0 + rr) * N_;

  // loop-invariant row constants in registers (LDS reloads blocked by barriers)
  float sR = sRow[rr], mR = mRow[rr], rI = rInv[rr];

  int col   = wv * 16 + (lane & 15);
  int jhalf = (lane >= 16) ? 8 : 0;

  v8f acc = {};
  int par = 0;
  for (int j0 = 0; j0 < N_; j0 += 32, par ^= 1) {
    // build 16x32 probability tile directly in A-fragment order (branchless)
    int j = j0 + jj;
    int2   av  = *(const int2*)&adjRow[j];
    float2 sdv = *(const float2*)&sd[j];
    __builtin_prefetch(&adjRow[j + 32], 0, 1);
    float e0 = (av.x > 0) ? leakyf(sR + sdv.x) : NEGV;
    float e1 = (av.y > 0) ? leakyf(sR + sdv.y) : NEGV;
    BfPair bp;
    bp.h[0] = (__bf16)(__expf(e0 - mR) * rI);
    bp.h[1] = (__bf16)(__expf(e1 - mR) * rI);
    pFragU[par][laneT * 8 + vT] = bp.u;
    __syncthreads();                 // single barrier/iter (double-buffered)

    Frag fa, fb;
    const uint4* pq = (const uint4*)&pFragU[par][lane * 8];
    fa.q[0] = pq[0];
    fa.q[1] = pq[1];
    const uint4* w4 = (const uint4*)(whSlc + (long)col * N_ + j0 + jhalf);
    fb.q[0] = w4[0];
    fb.q[1] = w4[2];
    acc = __builtin_amdgcn_wmma_f32_16x16x32_bf16(false, fa.v, false, fb.v,
                                                  (short)0, acc, false, false);
  }

#pragma unroll
  for (int i = 0; i < 8; ++i) {
    int r = i0 + i + ((lane >= 16) ? 8 : 0);
    float x = eluf(eluf(acc[i]));             // GAT elu + outer elu
    Hout[(rowbase + r) * H_ + col] = (__bf16)x;
  }
}

// ---------------- LSTM over T steps + output MLP -----------------------------
__global__ void __launch_bounds__(256)
k_lstm(const __bf16* __restrict__ G, const __bf16* __restrict__ WihP,
       const __bf16* __restrict__ WhhP,
       const float* __restrict__ b_ih, const float* __restrict__ b_hh,
       const float* __restrict__ Wo1, const float* __restrict__ bo1,
       const float* __restrict__ Wo2, const float* __restrict__ bo2,
       float* __restrict__ out) {
  __shared__ __align__(16) float hS[16][LH_];
  __shared__ __align__(16) float cS[16][LH_];
  __shared__ float gS[16][4 * LH_];

  int tid = threadIdx.x, lane = tid & 31, wv = tid >> 5;
  int s0 = blockIdx.x * 16;

  for (int idx = tid; idx < 16 * LH_; idx += 256) {
    (&hS[0][0])[idx] = 0.f;
    (&cS[0][0])[idx] = 0.f;
  }
  __syncthreads();

  int r   = lane & 15;
  int sA  = s0 + r;
  int bA  = sA >> 10, nA = sA & 1023;
  int klo = (lane >= 16) ? 8 : 0;
  int tn0 = 2 * wv, tn1 = 2 * wv + 1;             // gate col tiles (of 16)

  for (int t = 0; t < T_; ++t) {
    const __bf16* xrow = G + (((long)(bA * T_ + t)) * N_ + nA) * H_;
    v8f acc0 = {}, acc1 = {};

#pragma unroll
    for (int k0 = 0; k0 < H_; k0 += 32) {         // x_t @ W_ih^T
      Frag fa, f0, f1;
      const uint4* a4 = (const uint4*)(xrow + k0 + klo);
      fa.q[0] = a4[0];
      fa.q[1] = a4[2];
      const uint4* b0 = (const uint4*)(WihP + (((long)(k0 >> 5) * 16 + tn0) << 9) + (lane << 4));
      const uint4* b1 = (const uint4*)(WihP + (((long)(k0 >> 5) * 16 + tn1) << 9) + (lane << 4));
      f0.q[0] = b0[0]; f0.q[1] = b0[1];
      f1.q[0] = b1[0]; f1.q[1] = b1[1];
      acc0 = __builtin_amdgcn_wmma_f32_16x16x32_bf16(false, fa.v, false, f0.v, (short)0, acc0, false, false);
      acc1 = __builtin_amdgcn_wmma_f32_16x16x32_bf16(false, fa.v, false, f1.v, (short)0, acc1, false, false);
    }
#pragma unroll
    for (int k0 = 0; k0 < LH_; k0 += 32) {        // h @ W_hh^T
      Frag fa, f0, f1;
      const float4* h4 = (const float4*)&hS[r][k0 + klo];
      float4 u0 = h4[0], u1 = h4[1], u2 = h4[4], u3 = h4[5];
      float t16[16] = {u0.x,u0.y,u0.z,u0.w, u1.x,u1.y,u1.z,u1.w,
                       u2.x,u2.y,u2.z,u2.w, u3.x,u3.y,u3.z,u3.w};
#pragma unroll
      for (int e = 0; e < 16; ++e) fa.v[e] = (__bf16)t16[e];
      const uint4* b0 = (const uint4*)(WhhP + (((long)(k0 >> 5) * 16 + tn0) << 9) + (lane << 4));
      const uint4* b1 = (const uint4*)(WhhP + (((long)(k0 >> 5) * 16 + tn1) << 9) + (lane << 4));
      f0.q[0] = b0[0]; f0.q[1] = b0[1];
      f1.q[0] = b1[0]; f1.q[1] = b1[1];
      acc0 = __builtin_amdgcn_wmma_f32_16x16x32_bf16(false, fa.v, false, f0.v, (short)0, acc0, false, false);
      acc1 = __builtin_amdgcn_wmma_f32_16x16x32_bf16(false, fa.v, false, f1.v, (short)0, acc1, false, false);
    }

#pragma unroll
    for (int i = 0; i < 8; ++i) {                 // gates + biases -> LDS
      int rr  = i + ((lane >= 16) ? 8 : 0);
      int cc0 = tn0 * 16 + (lane & 15);
      int cc1 = tn1 * 16 + (lane & 15);
      gS[rr][cc0] = acc0[i] + b_ih[cc0] + b_hh[cc0];
      gS[rr][cc1] = acc1[i] + b_ih[cc1] + b_hh[cc1];
    }
    __syncthreads();

#pragma unroll
    for (int q = 0; q < 4; ++q) {                 // cell update (i,f,g,o)
      int idx = tid * 4 + q;
      int rr = idx >> 6, u = idx & 63;
      float gi = gS[rr][u], gf = gS[rr][64 + u], gg = gS[rr][128 + u], go = gS[rr][192 + u];
      float cc = sigmf(gf) * cS[rr][u] + sigmf(gi) * tanhf(gg);
      cS[rr][u] = cc;
      hS[rr][u] = sigmf(go) * tanhf(cc);
    }
    __syncthreads();
  }

  // MLP: relu(h @ Wo1 + bo1) @ Wo2 + bo2   (tmp in gS[16][32])
#pragma unroll
  for (int q = 0; q < 2; ++q) {
    int idx = tid * 2 + q;
    int rr = idx >> 5, mh = idx & 31;
    float s = bo1[mh];
#pragma unroll 8
    for (int k = 0; k < LH_; ++k) s += hS[rr][k] * Wo1[k * 32 + mh];
    gS[rr][mh] = (s > 0.f) ? s : 0.f;
  }
  __syncthreads();
  if (tid < 16) {
    float s = bo2[0];
#pragma unroll
    for (int m = 0; m < 32; ++m) s += gS[tid][m] * Wo2[m];
    out[s0 + tid] = s;
  }
}

// ---------------- host launcher ----------------------------------------------
extern "C" void kernel_launch(void* const* d_in, const int* in_sizes, int n_in,
                              void* d_out, int out_size, void* d_ws, size_t ws_size,
                              hipStream_t stream) {
  (void)in_sizes; (void)n_in; (void)out_size; (void)ws_size;

  const float* x    = (const float*)d_in[0];
  const int*   adj  = (const int*)  d_in[1];
  const float* W1   = (const float*)d_in[2];
  const float* a1   = (const float*)d_in[3];
  const float* W2   = (const float*)d_in[4];
  const float* a2   = (const float*)d_in[5];
  const float* W_ih = (const float*)d_in[6];
  const float* W_hh = (const float*)d_in[7];
  const float* b_ih = (const float*)d_in[8];
  const float* b_hh = (const float*)d_in[9];
  const float* Wo1  = (const float*)d_in[10];
  const float* bo1  = (const float*)d_in[11];
  const float* Wo2  = (const float*)d_in[12];
  const float* bo2  = (const float*)d_in[13];
  float* out = (float*)d_out;

  char* w = (char*)d_ws;
  auto carve = [&](size_t bytes) -> void* {
    void* p = (void*)w;
    w += (bytes + 255) & ~(size_t)255;
    return p;
  };
  __bf16* W1p  = (__bf16*)carve((size_t)FIN_ * H_ * 2);
  __bf16* W2p  = (__bf16*)carve((size_t)H_ * H_ * 2);
  __bf16* WihP = (__bf16*)carve((size_t)H_ * 4 * LH_ * 2);
  __bf16* WhhP = (__bf16*)carve((size_t)LH_ * 4 * LH_ * 2);
  __bf16* bufA = (__bf16*)carve((size_t)BTN_ * H_ * 2);   // WhT (transposed slices)
  __bf16* bufB = (__bf16*)carve((size_t)BTN_ * H_ * 2);   // layer activations, row-major
  float*  ssrc = (float*)carve((size_t)BTN_ * 4);
  float*  sdst = (float*)carve((size_t)BTN_ * 4);
  float*  rmax = (float*)carve((size_t)BTN_ * 4);
  float*  rsum = (float*)carve((size_t)BTN_ * 4);

  // weight prep (pack into B-fragment order, fused bf16 convert / transpose)
  k_pack_b<<<(FIN_ * H_ + 255) / 256, 256, 0, stream>>>(W1, W1p, FIN_, H_, H_, 0);
  k_pack_b<<<(H_ * H_ + 255) / 256, 256, 0, stream>>>(W2, W2p, H_, H_, H_, 0);
  k_pack_b<<<(H_ * 4 * LH_ + 255) / 256, 256, 0, stream>>>(W_ih, WihP, H_, 4 * LH_, H_, 1);
  k_pack_b<<<(LH_ * 4 * LH_ + 255) / 256, 256, 0, stream>>>(W_hh, WhhP, LH_, 4 * LH_, LH_, 1);

  const int gemm_blocks = (BTN_ / 16) * 4 / 8;             // 2048 (2 tiles/wave)
  const int attn_blocks = (B_ * T_) * (N_ / 16);           // 4096

  // ---- GAT layer 1 ----
  k_gemm<0, FIN_><<<gemm_blocks, 256, 0, stream>>>(x, W1p, bufA);
  k_scores<<<BTN_ / 256, 256, 0, stream>>>(bufA, a1, ssrc, sdst, BTN_);
  k_stats<<<BTN_ / 8, 256, 0, stream>>>(adj, ssrc, sdst, rmax, rsum);
  k_attn<<<attn_blocks, 256, 0, stream>>>(bufA, adj, ssrc, sdst, rmax, rsum, bufB);

  // ---- GAT layer 2 ----
  k_gemm<1, H_><<<gemm_blocks, 256, 0, stream>>>(bufB, W2p, bufA);
  k_scores<<<BTN_ / 256, 256, 0, stream>>>(bufA, a2, ssrc, sdst, BTN_);
  k_stats<<<BTN_ / 8, 256, 0, stream>>>(adj, ssrc, sdst, rmax, rsum);
  k_attn<<<attn_blocks, 256, 0, stream>>>(bufA, adj, ssrc, sdst, rmax, rsum, bufB);

  // ---- LSTM + MLP ----
  k_lstm<<<(B_ * N_) / 16, 256, 0, stream>>>(bufB, WihP, WhhP, b_ih, b_hh,
                                             Wo1, bo1, Wo2, bo2, out);
}
